// MoEShell_2869038154061
// MI455X (gfx1250) — matrix operand order repaired and checked by reading, not compiled, not run on hardware
//
#include <hip/hip_runtime.h>
#include <stdint.h>

// Problem constants (match reference)
#define T_ 2
#define N_ 4096
#define D_ 1024
#define E_ 8
#define O_ 1024
#define TOKS (T_ * N_)

// Tiling for the expert GEMM
#define BLK_TOK 32   // tokens per block (2 wave-rows of 16)
#define BLK_O   64   // outputs per block (4 wave-cols of 16)
#define KTILE   32   // K per WMMA step (bf16 16x16x32)
#define LDS_PAD 8    // pad rows to 40 halves (80B) -> 16B-aligned chunks

typedef __attribute__((ext_vector_type(16))) __bf16 v16bf;
typedef __attribute__((ext_vector_type(8)))  float  v8f;

// Workspace layout (bytes)
#define WS_GATES_OFF 0
#define WS_GATES_BYTES ((size_t)TOKS * E_ * 4)                       // 256 KB
#define WS_XB_OFF (WS_GATES_OFF + WS_GATES_BYTES)
#define WS_XB_BYTES ((size_t)TOKS * D_ * 2)                          // 16 MB
#define WS_WEB_OFF (WS_XB_OFF + WS_XB_BYTES)
#define WS_WEB_BYTES ((size_t)E_ * O_ * D_ * 2)                      // 16 MB
#define WS_REQUIRED (WS_WEB_OFF + WS_WEB_BYTES)

// ---------------------------------------------------------------------------
// Async global->LDS copies (gfx1250; tracked by ASYNCcnt).
// VDST takes the LDS byte offset: the LDS aperture truncates a flat shared
// address to addr[31:0], so the low dword of the generic pointer is valid.
// ---------------------------------------------------------------------------
__device__ __forceinline__ uint32_t lds_off(const void* p) {
  return (uint32_t)(uintptr_t)p;
}
__device__ __forceinline__ void async_copy_b128(uint32_t lds, const void* g) {
  asm volatile("global_load_async_to_lds_b128 %0, %1, off"
               :: "v"(lds), "v"(g) : "memory");
}
__device__ __forceinline__ void async_copy_b64(uint32_t lds, const void* g) {
  asm volatile("global_load_async_to_lds_b64 %0, %1, off"
               :: "v"(lds), "v"(g) : "memory");
}
__device__ __forceinline__ void wait_async0() {
  asm volatile("s_wait_asynccnt 0x0" ::: "memory");
}

// ---------------------------------------------------------------------------
// Kernel 1: gating.  One wave32 per token: logits[8] = x[token,:] @ Wg[t],
// top-2 (first index wins ties, like jax top_k), normalized gates written
// densely (zeros elsewhere) to gates[token][8].
// ---------------------------------------------------------------------------
__global__ __launch_bounds__(256) void moe_gates_kernel(
    const float* __restrict__ x, const float* __restrict__ wg,
    float* __restrict__ gates) {
  const int tok  = (int)((blockIdx.x * 256u + threadIdx.x) >> 5);
  const int lane = threadIdx.x & 31;
  if (tok >= TOKS) return;
  const int t = tok / N_;
  const float* __restrict__ xr = x + (size_t)tok * D_;
  const float* __restrict__ wt = wg + (size_t)t * D_ * E_;

  float part[E_];
#pragma unroll
  for (int e = 0; e < E_; ++e) part[e] = 0.0f;

  for (int d0 = lane * 4; d0 < D_; d0 += 32 * 4) {
    float4 xv = *(const float4*)(xr + d0);
    const float* xs = (const float*)&xv;
#pragma unroll
    for (int j = 0; j < 4; ++j) {
      const float* __restrict__ wr = wt + (size_t)(d0 + j) * E_;
      const float xj = xs[j];
#pragma unroll
      for (int e = 0; e < E_; ++e) part[e] += xj * wr[e];
    }
  }

#pragma unroll
  for (int off = 16; off > 0; off >>= 1) {
#pragma unroll
    for (int e = 0; e < E_; ++e) part[e] += __shfl_xor(part[e], off, 32);
  }

  if (lane == 0) {
    int   i0 = 0;
    float v0 = part[0];
#pragma unroll
    for (int e = 1; e < E_; ++e)
      if (part[e] > v0) { v0 = part[e]; i0 = e; }
    int   i1 = (i0 == 0) ? 1 : 0;
    float v1 = part[i1];
#pragma unroll
    for (int e = 0; e < E_; ++e)
      if (e != i0 && part[e] > v1) { v1 = part[e]; i1 = e; }
    const float inv = 1.0f / (v0 + v1 + 1e-9f);
    float g[E_];
#pragma unroll
    for (int e = 0; e < E_; ++e) g[e] = 0.0f;
    g[i0] = v0 * inv;
    g[i1] = v1 * inv;
    float* __restrict__ gr = gates + (size_t)tok * E_;
#pragma unroll
    for (int e = 0; e < E_; ++e) gr[e] = g[e];
  }
}

// ---------------------------------------------------------------------------
// One-time: x (f32, [tok][d]) -> xb (bf16, [tok][d]).  8 elems/thread.
// ---------------------------------------------------------------------------
__global__ __launch_bounds__(256) void x_convert_kernel(
    const float* __restrict__ x, __bf16* __restrict__ xb) {
  const size_t i = ((size_t)blockIdx.x * 256 + threadIdx.x) * 8;
  float4 a = *(const float4*)(x + i);
  float4 b = *(const float4*)(x + i + 4);
  const float* as = (const float*)&a;
  const float* bs = (const float*)&b;
#pragma unroll
  for (int j = 0; j < 4; ++j) xb[i + j] = (__bf16)as[j];
#pragma unroll
  for (int j = 0; j < 4; ++j) xb[i + 4 + j] = (__bf16)bs[j];
}

// ---------------------------------------------------------------------------
// One-time: We (f32, [e][d][o]) -> web (bf16, [e][o][d]) via 64x64 LDS tiles.
// ---------------------------------------------------------------------------
__global__ __launch_bounds__(256) void we_transpose_kernel(
    const float* __restrict__ we, __bf16* __restrict__ web) {
  __shared__ __bf16 sT[64][64 + 4];
  const int e  = blockIdx.x;
  const int d0 = blockIdx.y * 64;
  const int o0 = blockIdx.z * 64;
  const float* __restrict__ src = we + ((size_t)e * D_ + d0) * O_ + o0;

  const int r0 = threadIdx.x >> 4;   // 0..15
  const int cq = threadIdx.x & 15;   // 0..15 -> 4 floats along o
#pragma unroll
  for (int it = 0; it < 4; ++it) {
    const int r = r0 + it * 16;      // d row within tile
    float4 v = *(const float4*)(src + (size_t)r * O_ + cq * 4);
    const float* vs = (const float*)&v;
#pragma unroll
    for (int j = 0; j < 4; ++j) sT[r][cq * 4 + j] = (__bf16)vs[j];
  }
  __syncthreads();

  const int c  = threadIdx.x >> 2;   // o column 0..63
  const int rq = threadIdx.x & 3;    // d chunk of 16
  __bf16 tmp[16];
#pragma unroll
  for (int i = 0; i < 16; ++i) tmp[i] = sT[rq * 16 + i][c];
  __bf16* __restrict__ dst =
      web + ((size_t)e * O_ + o0 + c) * D_ + d0 + rq * 16;
#pragma unroll
  for (int i = 0; i < 16; ++i) dst[i] = tmp[i];
}

// ---------------------------------------------------------------------------
// Kernel 2 (fast path): dense per-expert GEMM, v_wmma_f32_16x16x32_bf16,
// tiles staged with global_load_async_to_lds, gates folded into the combine.
//   out[tok, o] = sum_e gates[tok,e] * sum_d x[tok,d] * We[e,d,o]
// Block = 256 threads (8 waves) -> 32 tokens x 64 outputs tile.
// ---------------------------------------------------------------------------
__global__ __launch_bounds__(256) void moe_expert_bf16_kernel(
    const __bf16* __restrict__ xb, const __bf16* __restrict__ web,
    const float* __restrict__ gates, float* __restrict__ out) {
  __shared__ __bf16 sA[BLK_TOK][KTILE + LDS_PAD];        // 32 x 40 halves
  __shared__ __bf16 sB[E_][BLK_O][KTILE + LDS_PAD];      // 8 x 64 x 40 halves
  __shared__ float  sG[BLK_TOK][E_];

  const int tid     = threadIdx.x;
  const int tokBase = blockIdx.x * BLK_TOK;
  const int oBase   = blockIdx.y * BLK_O;

  const int lane  = tid & 31;
  const int wave  = tid >> 5;
  const int wr    = wave >> 2;     // wave row (0..1) -> 16 tokens
  const int wc    = wave & 3;      // wave col (0..3) -> 16 outputs
  const int nloc  = lane & 15;
  const int khalf = lane >> 4;
  const int mhi   = khalf * 8;

  sG[tid >> 3][tid & 7] = gates[(size_t)(tokBase + (tid >> 3)) * E_ + (tid & 7)];

  // Staging roles (constant over the K loop)
  const int a_tok = tid >> 3;      // 0..31
  const int a_kq  = tid & 7;       // 0..7 -> 4 halves (8B)
  const int b_e   = tid >> 5;      // expert = wave id
  const int b_lid = tid & 31;

  v8f acc[E_];
#pragma unroll
  for (int e = 0; e < E_; ++e)
#pragma unroll
    for (int i = 0; i < 8; ++i) acc[e][i] = 0.0f;

  for (int dk = 0; dk < D_ / KTILE; ++dk) {
    __syncthreads();  // previous compute done (also covers sG on iter 0)

    // ---- stage A: xb[tokBase..+32][dk*32..+32], async b64 per thread ----
    async_copy_b64(lds_off(&sA[a_tok][a_kq * 4]),
                   xb + (size_t)(tokBase + a_tok) * D_ + dk * KTILE + a_kq * 4);

    // ---- stage B: web[e][oBase..+64][dk*32..+32]; rows are contiguous in
    //      k thanks to the [e][o][d] pre-transpose -> 8 async b128/thread ----
#pragma unroll
    for (int rr = 0; rr < 2; ++rr) {
      const int oo = b_lid * 2 + rr;  // 0..63
      const __bf16* srcrow =
          web + ((size_t)b_e * O_ + oBase + oo) * D_ + dk * KTILE;
#pragma unroll
      for (int cq = 0; cq < 4; ++cq)
        async_copy_b128(lds_off(&sB[b_e][oo][cq * 8]), srcrow + cq * 8);
    }

    wait_async0();
    __syncthreads();

    // ---- A fragment (contiguous k, same as B: lanes<16 K{0..15}, else
    //      K{16..31}; M = lane&15) -- shared across all experts ----
    v16bf af;
    {
      const __bf16* __restrict__ ap = &sA[wr * 16 + nloc][0];
      const int kb = khalf * 16;
#pragma unroll
      for (int i = 0; i < 8; ++i) af[i] = ap[kb + i];
#pragma unroll
      for (int i = 0; i < 8; ++i) af[8 + i] = ap[kb + 8 + i];
    }

    // ---- one WMMA per expert ----
#pragma unroll
    for (int e = 0; e < E_; ++e) {
      v16bf bf_;
      const __bf16* __restrict__ bp = &sB[e][wc * 16 + nloc][khalf * 16];
#pragma unroll
      for (int i = 0; i < 16; ++i) bf_[i] = bp[i];
      acc[e] = __builtin_amdgcn_wmma_f32_16x16x32_bf16(
          false, af, false, bf_, (short)0, acc[e], false, false);
    }
  }

  float outv[8];
#pragma unroll
  for (int r = 0; r < 8; ++r) outv[r] = 0.0f;
#pragma unroll
  for (int e = 0; e < E_; ++e)
#pragma unroll
    for (int r = 0; r < 8; ++r)
      outv[r] += sG[wr * 16 + mhi + r][e] * acc[e][r];
#pragma unroll
  for (int r = 0; r < 8; ++r) {
    const size_t tokG = (size_t)(tokBase + wr * 16 + mhi + r);
    out[tokG * O_ + oBase + wc * 16 + nloc] = outv[r];
  }
}

// ---------------------------------------------------------------------------
// Kernel 2 (fallback, used only if ws is too small for the bf16 copies):
// converts f32 tiles to bf16 in-flight (round-1 version).
// ---------------------------------------------------------------------------
__global__ __launch_bounds__(256) void moe_expert_f32src_kernel(
    const float* __restrict__ x, const float* __restrict__ we,
    const float* __restrict__ gates, float* __restrict__ out) {
  __shared__ __bf16 sA[BLK_TOK][KTILE + LDS_PAD];
  __shared__ __bf16 sB[E_][BLK_O][KTILE + LDS_PAD];
  __shared__ float  sG[BLK_TOK][E_];

  const int tid     = threadIdx.x;
  const int tokBase = blockIdx.x * BLK_TOK;
  const int oBase   = blockIdx.y * BLK_O;
  const int lane  = tid & 31;
  const int wave  = tid >> 5;
  const int wr    = wave >> 2;
  const int wc    = wave & 3;
  const int nloc  = lane & 15;
  const int khalf = lane >> 4;
  const int mhi   = khalf * 8;

  sG[tid >> 3][tid & 7] = gates[(size_t)(tokBase + (tid >> 3)) * E_ + (tid & 7)];

  v8f acc[E_];
#pragma unroll
  for (int e = 0; e < E_; ++e)
#pragma unroll
    for (int i = 0; i < 8; ++i) acc[e][i] = 0.0f;

  for (int dk = 0; dk < D_ / KTILE; ++dk) {
    __syncthreads();
    {
      const int tok = tid >> 3, kq = tid & 7;
      float4 xv = *(const float4*)(x + (size_t)(tokBase + tok) * D_ +
                                   dk * KTILE + kq * 4);
      const float* xs = (const float*)&xv;
#pragma unroll
      for (int j = 0; j < 4; ++j) sA[tok][kq * 4 + j] = (__bf16)xs[j];
    }
    {
      const int e = tid >> 5, lid = tid & 31;
      const float* __restrict__ wb =
          we + ((size_t)e * D_ + (size_t)dk * KTILE) * O_ + oBase;
      for (int it = 0; it < 16; ++it) {
        const int li = it * 32 + lid;
        const int kk = li >> 4, oq = li & 15;
        float4 wv = *(const float4*)(wb + (size_t)kk * O_ + oq * 4);
        const float* ws = (const float*)&wv;
#pragma unroll
        for (int j = 0; j < 4; ++j) sB[e][oq * 4 + j][kk] = (__bf16)ws[j];
      }
    }
    __syncthreads();

    v16bf af;
    {
      const __bf16* __restrict__ ap = &sA[wr * 16 + nloc][0];
      const int kb = khalf * 16;
#pragma unroll
      for (int i = 0; i < 8; ++i) af[i] = ap[kb + i];
#pragma unroll
      for (int i = 0; i < 8; ++i) af[8 + i] = ap[kb + 8 + i];
    }
#pragma unroll
    for (int e = 0; e < E_; ++e) {
      v16bf bf_;
      const __bf16* __restrict__ bp = &sB[e][wc * 16 + nloc][khalf * 16];
#pragma unroll
      for (int i = 0; i < 16; ++i) bf_[i] = bp[i];
      acc[e] = __builtin_amdgcn_wmma_f32_16x16x32_bf16(
          false, af, false, bf_, (short)0, acc[e], false, false);
    }
  }

  float outv[8];
#pragma unroll
  for (int r = 0; r < 8; ++r) outv[r] = 0.0f;
#pragma unroll
  for (int e = 0; e < E_; ++e)
#pragma unroll
    for (int r = 0; r < 8; ++r)
      outv[r] += sG[wr * 16 + mhi + r][e] * acc[e][r];
#pragma unroll
  for (int r = 0; r < 8; ++r) {
    const size_t tokG = (size_t)(tokBase + wr * 16 + mhi + r);
    out[tokG * O_ + oBase + wc * 16 + nloc] = outv[r];
  }
}

// ---------------------------------------------------------------------------
extern "C" void kernel_launch(void* const* d_in, const int* in_sizes, int n_in,
                              void* d_out, int out_size, void* d_ws,
                              size_t ws_size, hipStream_t stream) {
  (void)in_sizes; (void)n_in; (void)out_size;
  const float* x  = (const float*)d_in[0];  // [T,N,D] f32
  const float* wg = (const float*)d_in[1];  // [T,D,E] f32
  const float* we = (const float*)d_in[2];  // [E,D,O] f32
  float* out   = (float*)d_out;             // [T,N,O] f32
  char*  ws    = (char*)d_ws;
  float* gates = (float*)(ws + WS_GATES_OFF);

  moe_gates_kernel<<<dim3(TOKS * 32 / 256), dim3(256), 0, stream>>>(x, wg,
                                                                    gates);

  if (ws_size >= WS_REQUIRED) {
    __bf16* xb  = (__bf16*)(ws + WS_XB_OFF);
    __bf16* web = (__bf16*)(ws + WS_WEB_OFF);
    x_convert_kernel<<<dim3((TOKS * D_) / (256 * 8)), dim3(256), 0, stream>>>(
        x, xb);
    we_transpose_kernel<<<dim3(E_, D_ / 64, O_ / 64), dim3(256), 0, stream>>>(
        we, web);
    moe_expert_bf16_kernel<<<dim3(TOKS / BLK_TOK, O_ / BLK_O), dim3(256), 0,
                             stream>>>(xb, web, gates, out);
  } else {
    moe_expert_f32src_kernel<<<dim3(TOKS / BLK_TOK, O_ / BLK_O), dim3(256), 0,
                               stream>>>(x, we, gates, out);
  }
}
